// AGCRN_29265907155020
// MI455X (gfx1250) — matrix operand order repaired and checked by reading, not compiled, not run on hardware
//
#include <hip/hip_runtime.h>
#include <hip/hip_bf16.h>

typedef __attribute__((ext_vector_type(16))) __bf16 v16bf;
typedef __attribute__((ext_vector_type(8)))  float  v8f;

#define THREADS 256

__device__ __forceinline__ float fast_sigmoid(float x) {
  // rcp(1 + exp(-x)): saturates correctly at both ends (exp->inf => rcp(inf)=0)
  return __builtin_amdgcn_rcpf(1.f + __expf(-x));
}

__device__ __forceinline__ float fast_tanh(float x) {
  // tanh(x) = (e^2x - 1) / (e^2x + 1), clamp so e^2x stays finite
  const float xc = fminf(fmaxf(x, -15.f), 15.f);
  const float t = __expf(2.f * xc);
  return (t - 1.f) * __builtin_amdgcn_rcpf(t + 1.f);
}

// ---------------- one-time setup kernels ----------------

__global__ __launch_bounds__(THREADS) void init_node_kernel(
    float* __restrict__ dmax, float* __restrict__ denom, float* __restrict__ deg, int NT) {
  int i = blockIdx.x * blockDim.x + threadIdx.x;
  if (i < NT) { dmax[i] = 0.f; denom[i] = 0.f; deg[i] = 1.f; /* self loop weight */ }
}

__global__ __launch_bounds__(THREADS) void zero_kernel(float* __restrict__ p, int n) {
  int i = blockIdx.x * blockDim.x + threadIdx.x;
  if (i < n) p[i] = 0.f;
}

__global__ __launch_bounds__(THREADS) void cvt_bf16_kernel(
    const float* __restrict__ a, const float* __restrict__ b,
    __bf16* __restrict__ ab, __bf16* __restrict__ bb, int n) {
  int i = blockIdx.x * blockDim.x + threadIdx.x;
  if (i < n) { ab[i] = (__bf16)a[i]; bb[i] = (__bf16)b[i]; }
}

// dots = relu(<emb[src], emb[tgt]>); dmax[src] = max(dmax[src], dots)   (dots >= 0, dmax init 0)
__global__ __launch_bounds__(THREADS) void edge_dots_kernel(
    const int* __restrict__ src, const int* __restrict__ tgt,
    const float* __restrict__ emb, float* __restrict__ dots,
    unsigned* __restrict__ dmax, int E, int ED) {
  int e = blockIdx.x * blockDim.x + threadIdx.x;
  if (e >= E) return;
  const int s = src[e], t = tgt[e];
  const float* es = emb + (size_t)s * ED;
  const float* et = emb + (size_t)t * ED;
  float d = 0.f;
  for (int i = 0; i < ED; ++i) d += es[i] * et[i];
  d = fmaxf(d, 0.f);
  dots[e] = d;
  atomicMax(&dmax[s], __float_as_uint(d));  // valid ordering for non-negative floats
}

// w = exp(dots - dmax[src]); denom[src] += w
__global__ __launch_bounds__(THREADS) void edge_w_kernel(
    const int* __restrict__ src, const float* __restrict__ dots,
    const float* __restrict__ dmax, float* __restrict__ w,
    float* __restrict__ denom, int E) {
  int e = blockIdx.x * blockDim.x + threadIdx.x;
  if (e >= E) return;
  const int s = src[e];
  const float wv = __expf(dots[e] - dmax[s]);
  w[e] = wv;
  atomicAdd(&denom[s], wv);
}

// edge_weight = w/(denom[src]+1e-8); deg[tgt] += edge_weight  (deg init 1 = self loop)
__global__ __launch_bounds__(THREADS) void edge_ew_kernel(
    const int* __restrict__ src, const int* __restrict__ tgt,
    const float* __restrict__ w, const float* __restrict__ denom,
    float* __restrict__ nrm, float* __restrict__ deg, int E) {
  int e = blockIdx.x * blockDim.x + threadIdx.x;
  if (e >= E) return;
  const int s = src[e], t = tgt[e];
  const float ew = w[e] / (denom[s] + 1e-8f);
  nrm[e] = ew;
  atomicAdd(&deg[t], ew);
}

__global__ __launch_bounds__(THREADS) void dinv_kernel(
    const float* __restrict__ deg, float* __restrict__ dinv, int NT) {
  int i = blockIdx.x * blockDim.x + threadIdx.x;
  if (i < NT) { float d = deg[i]; dinv[i] = d > 0.f ? rsqrtf(d) : 0.f; }
}

// nrm[e] = dinv[src]*edge_weight*dinv[tgt]
__global__ __launch_bounds__(THREADS) void norm_kernel(
    const int* __restrict__ src, const int* __restrict__ tgt,
    const float* __restrict__ dinv, float* __restrict__ nrm, int E) {
  int e = blockIdx.x * blockDim.x + threadIdx.x;
  if (e >= E) return;
  nrm[e] = dinv[src[e]] * nrm[e] * dinv[tgt[e]];
}

// Y[t*NT+i] = dinv[i]^2 * x[i*T+t]   (self-loop contribution of scalar propagation)
__global__ __launch_bounds__(THREADS) void y_init_kernel(
    const float* __restrict__ x, const float* __restrict__ dinv,
    float* __restrict__ Y, int NT, int T) {
  int idx = blockIdx.x * blockDim.x + threadIdx.x;
  if (idx >= NT * T) return;
  const int t = idx / NT, i = idx - t * NT;
  const float di = dinv[i];
  Y[idx] = di * di * x[(size_t)i * T + t];
}

// Y[t*NT+tgt] += nrm * x[src*T+t] for all 24 t (one edge read, 24 FMAs + atomics)
__global__ __launch_bounds__(THREADS) void y_scatter_kernel(
    const int* __restrict__ src, const int* __restrict__ tgt,
    const float* __restrict__ nrm, const float* __restrict__ x,
    float* __restrict__ Y, int E, int NT) {
  int e = blockIdx.x * blockDim.x + threadIdx.x;
  if (e >= E) return;
  const int s = src[e], d = tgt[e];
  const float c = nrm[e];
  const float4* xp = (const float4*)(x + (size_t)s * 24);
  float xv[24];
#pragma unroll
  for (int q = 0; q < 6; ++q) {
    float4 v = xp[q];
    xv[4 * q + 0] = v.x; xv[4 * q + 1] = v.y; xv[4 * q + 2] = v.z; xv[4 * q + 3] = v.w;
  }
#pragma unroll
  for (int t = 0; t < 24; ++t) atomicAdd(&Y[(size_t)t * NT + d], c * xv[t]);
}

// ---------------- GRU step: the WMMA workhorse ----------------
// Per block: 16 rows. 48 output tiles of 16x16 (24 for gi = xt@Wih^T, 24 for gh = h@Whh^T),
// 8 waves x 6 tiles each, K = 128 -> 4 x v_wmma_f32_16x16x32_bf16 per tile. Bias in C operand.

__global__ __launch_bounds__(THREADS) void gru_step_kernel(
    const float* __restrict__ Yt, const float* __restrict__ hin,
    float* __restrict__ hout,
    const __bf16* __restrict__ Wih, const __bf16* __restrict__ Whh,
    const float* __restrict__ Win, const float* __restrict__ bin,
    const float* __restrict__ bih, const float* __restrict__ bhh, int NT) {
  __shared__ __bf16 sXt[16][136];     // xt tile (A for gi), padded vs bank conflicts
  __shared__ __bf16 sH[16][136];      // h  tile (A for gh)
  __shared__ float  sG[2][16][388];   // [0]=gi, [1]=gh, full 384 cols per row
  __shared__ float  sY[16];

  const int tid = threadIdx.x;
  const int row0 = blockIdx.x << 4;

  if (tid < 16) {
    const int r = row0 + tid;
    sY[tid] = (r < NT) ? Yt[r] : 0.f;
  }
  __syncthreads();

  for (int idx = tid; idx < 16 * 128; idx += THREADS) {
    const int m = idx >> 7, k = idx & 127;
    const int r = row0 + m;
    const float hv = (r < NT) ? hin[(size_t)r * 128 + k] : 0.f;
    sH[m][k] = (__bf16)hv;
    sXt[m][k] = (__bf16)fmaxf(sY[m] * Win[k] + bin[k], 0.f);  // relu(GCNConv 1->128)
  }
  __syncthreads();

  const int lane = tid & 31;
  const int wave = tid >> 5;
  const int nloc = lane & 15;        // C/D: column; B: column; A: row
  const int hi   = lane >> 4;
  const int kbA  = hi << 3;          // unused: A is loaded with kbB, same k order as B
  const int kbB  = hi << 4;          // A and B (contiguous k order): lane>=16 holds K 16..31
  const int moff = hi << 3;          // C/D: lanes>=16 hold M=r+8

  for (int i = 0; i < 6; ++i) {
    const int tt = wave + (i << 3);           // 0..47
    const int which = (tt >= 24) ? 1 : 0;     // 0: gi, 1: gh
    const int nt = which ? (tt - 24) : tt;    // 16-col tile within 384
    const __bf16* __restrict__ W = which ? Whh : Wih;            // [384][128] row-major
    const __bf16 (*A)[136] = which ? sH : sXt;
    const int cg = nt * 16 + nloc;            // global output column (0..383)
    const float bv = (which ? bhh : bih)[cg];

    v8f acc;
#pragma unroll
    for (int r = 0; r < 8; ++r) acc[r] = bv;  // bias broadcast across M

#pragma unroll
    for (int ks = 0; ks < 4; ++ks) {
      const int k0 = ks << 5;
      v16bf a, b;
#pragma unroll
      for (int j = 0; j < 8; ++j) {
        const int ka = k0 + kbB + 2 * j;
        const int kb = k0 + kbB + 2 * j;
        a[2 * j]     = A[nloc][ka];
        a[2 * j + 1] = A[nloc][ka + 1];
        b[2 * j]     = W[(size_t)cg * 128 + kb];       // B[k][n] = W[n][k]
        b[2 * j + 1] = W[(size_t)cg * 128 + kb + 1];
      }
      acc = __builtin_amdgcn_wmma_f32_16x16x32_bf16(
          false, a, false, b, (short)0, acc, false, false);
    }
#pragma unroll
    for (int r = 0; r < 8; ++r) sG[which][moff + r][cg] = acc[r];
  }
  __syncthreads();

  // GRU gate math + write h_out (fast rcp-based sigmoid/tanh, no precise-div chains)
  for (int idx = tid; idx < 16 * 128; idx += THREADS) {
    const int m = idx >> 7, k = idx & 127;
    const int r = row0 + m;
    if (r >= NT) continue;
    const float ir = sG[0][m][k],       hr = sG[1][m][k];
    const float iz = sG[0][m][128 + k], hz = sG[1][m][128 + k];
    const float ig = sG[0][m][256 + k], hg = sG[1][m][256 + k];
    const float rg = fast_sigmoid(ir + hr);
    const float zg = fast_sigmoid(iz + hz);
    const float ng = fast_tanh(ig + rg * hg);
    const float hv = hin[(size_t)r * 128 + k];
    hout[(size_t)r * 128 + k] = (1.f - zg) * ng + zg * hv;
  }
}

// ---------------- output layer (scalar propagation again) ----------------

__global__ __launch_bounds__(THREADS) void out_init_kernel(
    const float* __restrict__ h, const float* __restrict__ Wout,
    const float* __restrict__ bout, const float* __restrict__ dinv,
    float* __restrict__ p, float* __restrict__ out, int NT) {
  int i = blockIdx.x * blockDim.x + threadIdx.x;
  if (i >= NT) return;
  const float* hr = h + (size_t)i * 128;
  float pv = 0.f;
#pragma unroll 8
  for (int k = 0; k < 128; ++k) pv += hr[k] * Wout[k];
  p[i] = pv;
  const float di = dinv[i];
  out[i] = di * di * pv + bout[0];
}

__global__ __launch_bounds__(THREADS) void out_scatter_kernel(
    const int* __restrict__ src, const int* __restrict__ tgt,
    const float* __restrict__ nrm, const float* __restrict__ p,
    float* __restrict__ out, int E) {
  int e = blockIdx.x * blockDim.x + threadIdx.x;
  if (e >= E) return;
  atomicAdd(&out[tgt[e]], nrm[e] * p[src[e]]);
}

// ---------------- host launcher ----------------

extern "C" void kernel_launch(void* const* d_in, const int* in_sizes, int n_in,
                              void* d_out, int out_size, void* d_ws, size_t ws_size,
                              hipStream_t stream) {
  const float* x_seq = (const float*)d_in[0];
  const int*   eidx  = (const int*)d_in[1];
  const float* emb   = (const float*)d_in[2];
  const float* W_in  = (const float*)d_in[3];
  const float* b_in  = (const float*)d_in[4];
  const float* W_ih  = (const float*)d_in[5];
  const float* W_hh  = (const float*)d_in[6];
  const float* b_ih  = (const float*)d_in[7];
  const float* b_hh  = (const float*)d_in[8];
  const float* W_out = (const float*)d_in[9];
  const float* b_out = (const float*)d_in[10];

  const int T   = 24;
  const int HC  = 128;
  const int HG3 = 384;
  const int NT  = in_sizes[0] / T;       // 20000
  const int E   = in_sizes[1] / 2;       // 640000
  const int ED  = in_sizes[2] / NT;      // 10
  const int* src = eidx;
  const int* tgt = eidx + E;

  // workspace carve-out (~31 MB)
  char* ws = (char*)d_ws;
  size_t off = 0;
  auto carve = [&](size_t bytes) -> void* {
    void* pp = ws + off;
    off += (bytes + 255) & ~(size_t)255;
    return pp;
  };
  float*  d_dots  = (float*)carve((size_t)E * 4);
  float*  d_dmax  = (float*)carve((size_t)NT * 4);
  float*  d_w     = (float*)carve((size_t)E * 4);
  float*  d_denom = (float*)carve((size_t)NT * 4);
  float*  d_deg   = (float*)carve((size_t)NT * 4);
  float*  d_dinv  = (float*)carve((size_t)NT * 4);
  float*  d_nrm   = (float*)carve((size_t)E * 4);
  float*  d_Y     = (float*)carve((size_t)T * NT * 4);
  __bf16* d_Wihb  = (__bf16*)carve((size_t)HG3 * HC * 2);
  __bf16* d_Whhb  = (__bf16*)carve((size_t)HG3 * HC * 2);
  float*  d_hA    = (float*)carve((size_t)NT * HC * 4);
  float*  d_hB    = (float*)carve((size_t)NT * HC * 4);
  float*  d_p     = (float*)carve((size_t)NT * 4);

  const int TB = THREADS;
  const int gN = (NT + TB - 1) / TB;
  const int gE = (E + TB - 1) / TB;

  // setup: edge softmax + gcn norm (time-invariant)
  init_node_kernel<<<gN, TB, 0, stream>>>(d_dmax, d_denom, d_deg, NT);
  zero_kernel<<<(NT * HC + TB - 1) / TB, TB, 0, stream>>>(d_hA, NT * HC);
  cvt_bf16_kernel<<<(HG3 * HC + TB - 1) / TB, TB, 0, stream>>>(W_ih, W_hh, d_Wihb, d_Whhb, HG3 * HC);
  edge_dots_kernel<<<gE, TB, 0, stream>>>(src, tgt, emb, d_dots, (unsigned*)d_dmax, E, ED);
  edge_w_kernel<<<gE, TB, 0, stream>>>(src, d_dots, d_dmax, d_w, d_denom, E);
  edge_ew_kernel<<<gE, TB, 0, stream>>>(src, tgt, d_w, d_denom, d_nrm, d_deg, E);
  dinv_kernel<<<gN, TB, 0, stream>>>(d_deg, d_dinv, NT);
  norm_kernel<<<gE, TB, 0, stream>>>(src, tgt, d_dinv, d_nrm, E);

  // scalar graph propagation of all 24 timesteps in one pass
  y_init_kernel<<<(NT * T + TB - 1) / TB, TB, 0, stream>>>(x_seq, d_dinv, d_Y, NT, T);
  y_scatter_kernel<<<gE, TB, 0, stream>>>(src, tgt, d_nrm, x_seq, d_Y, E, NT);

  // 24 GRU steps via bf16 WMMA
  float* hcur = d_hA;
  float* hnxt = d_hB;
  for (int t = 0; t < T; ++t) {
    gru_step_kernel<<<(NT + 15) / 16, TB, 0, stream>>>(
        d_Y + (size_t)t * NT, hcur, hnxt, d_Wihb, d_Whhb, W_in, b_in, b_ih, b_hh, NT);
    float* tmp = hcur; hcur = hnxt; hnxt = tmp;
  }

  // output GCNConv(128->1): scalar propagation of p = h . W_out
  float* out = (float*)d_out;
  out_init_kernel<<<gN, TB, 0, stream>>>(hcur, W_out, b_out, d_dinv, d_p, out, NT);
  out_scatter_kernel<<<gE, TB, 0, stream>>>(src, tgt, d_nrm, d_p, out, E);
}
